// GraphAttnEmbedding_18545668784187
// MI455X (gfx1250) — matrix operand order repaired and checked by reading, not compiled, not run on hardware
//
#include <hip/hip_runtime.h>
#include <math.h>
#include <stdint.h>

#define IN_CH    128
#define OUT_CH   128
#define HEADS    4
#define HEAD_DIM 32
#define TIME_DIM 100

typedef __attribute__((ext_vector_type(16))) __bf16 v16bf;
typedef __attribute__((ext_vector_type(8)))  __bf16 v8bf;
typedef __attribute__((ext_vector_type(8)))  float  v8f;
typedef __attribute__((ext_vector_type(4)))  unsigned int v4u;
typedef __attribute__((ext_vector_type(8)))  unsigned int v8u;

// ---------- helpers -------------------------------------------------------

__device__ __forceinline__ unsigned short f2bf(float f) {
  unsigned int u = __float_as_uint(f);
  u += 0x7FFFu + ((u >> 16) & 1u);          // round-to-nearest-even
  return (unsigned short)(u >> 16);
}

// A-matrix fragment: 16x32 bf16, row-major tile (ld in elements).
// ISA 7.12.2: lane L holds row M=L&15; lanes<16 get K[0..7],[16..23];
// lanes>=16 get K[8..15],[24..31].
__device__ __forceinline__ v16bf load_a_frag(const unsigned short* tile, int ld,
                                             int lane, int k0) {
  const int m  = lane & 15;
  const int kb = k0 + ((lane & 16) ? 8 : 0);
  const unsigned short* p = tile + m * ld + kb;
  v8bf lo = *(const v8bf*)(p);
  v8bf hi = *(const v8bf*)(p + 16);
  return __builtin_shufflevector(lo, hi, 0,1,2,3,4,5,6,7,8,9,10,11,12,13,14,15);
}

// B-matrix fragment: 32x16 bf16. WT is pre-transposed [out=128][in=128] so
// each lane (column n) reads K like A: lanes<16 K[0..7],[16..23],
// lanes>=16 K[8..15],[24..31].
__device__ __forceinline__ v16bf load_b_frag(const unsigned short* WT, int n0,
                                             int lane, int k0) {
  const int n  = n0 + (lane & 15);
  const int kb = k0 + ((lane & 16) ? 8 : 0);
  const unsigned short* p = WT + n * IN_CH + kb;
  v8bf lo = *(const v8bf*)(p);
  v8bf hi = *(const v8bf*)(p + 16);
  return __builtin_shufflevector(lo, hi, 0,1,2,3,4,5,6,7,8,9,10,11,12,13,14,15);
}

__device__ __forceinline__ v8f wmma_bf16(v16bf a, v16bf b, v8f c) {
  return __builtin_amdgcn_wmma_f32_16x16x32_bf16(false, a, false, b,
                                                 (short)0, c, false, false);
}

// C/D layout: element r -> row = r + 8*(lane>=16), col = lane&15
__device__ __forceinline__ void store_tile_f32(float* out, int ld, const v8f& acc,
                                               const float* bias, int n0, int lane) {
  const int col = n0 + (lane & 15);
  const int rb  = (lane & 16) ? 8 : 0;
  const float b = bias ? bias[col] : 0.0f;
#pragma unroll
  for (int r = 0; r < 8; ++r)
    out[(size_t)(rb + r) * ld + col] = acc[r] + b;
}

__device__ __forceinline__ void atomicMaxF(float* addr, float v) {
  if (v >= 0.0f) atomicMax((int*)addr, __float_as_int(v));
  else           atomicMin((unsigned int*)addr, __float_as_uint(v));
}

// ---------- TDM gather: 16 scattered rows (128 f32 each) -> LDS ----------
// ISA ch.8: gather-mode tensor descriptor, 16-bit row indices (N < 65536),
// data_size=4B, tile = 16 x 128, row stride 128 elements. Issued once by a
// single wave; overlapped with edge-attr build + WMMA, consumed after
// s_wait_tensorcnt + barrier.
__device__ __forceinline__ void tdm_gather16(const float* __restrict__ base,
                                             unsigned lds_off,
                                             const int* __restrict__ idx, int i0,
                                             unsigned dim1) {
  unsigned r[16];
#pragma unroll
  for (int i = 0; i < 16; ++i)
    r[i] = (unsigned)__builtin_amdgcn_readfirstlane(idx[i0 + i]) & 0xffffu;

  const unsigned long long ga = (unsigned long long)(uintptr_t)base;
  // group0: count=1, gather_mode=1 (bit31), 16-bit indices (bit30=0),
  //         lds_addr, 57-bit global byte address, type=2 (bits 127:126)
  v4u g0 = { 0x80000001u,
             lds_off,
             (unsigned)ga,
             (unsigned)(ga >> 32) | (2u << 30) };
  // group1: data_size=2 (4B); tensor_dim0=128; tensor_dim1=dim1;
  //         tile_dim0=128; tile_dim1=16 (#indices); tensor_dim0_stride=128
  v8u g1 = { 2u << 16,
             128u << 16,
             (dim1 & 0xffffu) << 16,
             ((dim1 >> 16) & 0xffffu) | (128u << 16),
             16u,
             128u,
             0u, 0u };
  // groups 2/3: sixteen packed 16-bit row indices
  v4u g2 = { r[0] | (r[1] << 16),  r[2]  | (r[3] << 16),
             r[4] | (r[5] << 16),  r[6]  | (r[7] << 16) };
  v4u g3 = { r[8] | (r[9] << 16),  r[10] | (r[11] << 16),
             r[12] | (r[13] << 16), r[14] | (r[15] << 16) };
  asm volatile("tensor_load_to_lds %0, %1, %2, %3"
               :: "s"(g0), "s"(g1), "s"(g2), "s"(g3) : "memory");
}

// ---------- conversion / init kernels ------------------------------------

__global__ void conv_x_kernel(const float* __restrict__ x,
                              unsigned short* __restrict__ xb, int n) {
  int i = blockIdx.x * blockDim.x + threadIdx.x;
  if (i < n) xb[i] = f2bf(x[i]);
}

// Transpose + bf16-convert the 5 weight matrices; We's K (101) padded to 128.
__global__ void conv_w_kernel(const float* __restrict__ Wq, const float* __restrict__ Wk,
                              const float* __restrict__ Wv, const float* __restrict__ Ws,
                              const float* __restrict__ We,
                              unsigned short* __restrict__ WqT, unsigned short* __restrict__ WkT,
                              unsigned short* __restrict__ WvT, unsigned short* __restrict__ WsT,
                              unsigned short* __restrict__ WeT) {
  int idx = blockIdx.x * blockDim.x + threadIdx.x;
  if (idx >= 5 * 128 * 128) return;
  int wsel = idx >> 14;
  int rem  = idx & 16383;
  int o = rem >> 7, i = rem & 127;
  float v; unsigned short* dst;
  switch (wsel) {
    case 0:  v = Wq[i * 128 + o]; dst = WqT; break;
    case 1:  v = Wk[i * 128 + o]; dst = WkT; break;
    case 2:  v = Wv[i * 128 + o]; dst = WvT; break;
    case 3:  v = Ws[i * 128 + o]; dst = WsT; break;
    default: v = (i < TIME_DIM + 1) ? We[i * 128 + o] : 0.0f; dst = WeT; break;
  }
  dst[rem] = f2bf(v);
}

__global__ void init_kernel(float* __restrict__ agg, float* __restrict__ m,
                            float* __restrict__ den, int N) {
  int i = blockIdx.x * blockDim.x + threadIdx.x;
  const int nagg = N * OUT_CH, nh = N * HEADS;
  if (i < nagg)                 agg[i] = 0.0f;
  else if (i < nagg + nh)       m[i - nagg] = -__builtin_huge_valf();
  else if (i < nagg + 2 * nh)   den[i - nagg - nh] = 0.0f;
}

// ---------- node projections: Q/K/V/Skip via WMMA -------------------------

__global__ __launch_bounds__(256)
void node_proj_kernel(const unsigned short* __restrict__ xb,
                      const unsigned short* __restrict__ WqT, const unsigned short* __restrict__ WkT,
                      const unsigned short* __restrict__ WvT, const unsigned short* __restrict__ WsT,
                      const float* __restrict__ bq, const float* __restrict__ bk,
                      const float* __restrict__ bv, const float* __restrict__ bs,
                      float* __restrict__ Q, float* __restrict__ K,
                      float* __restrict__ V, float* __restrict__ S) {
  const int rt   = blockIdx.x;          // 16-row tile
  const int wave = threadIdx.x >> 5;    // 8 waves -> 8 col tiles (128 cols)
  const int lane = threadIdx.x & 31;
  const int n0   = wave * 16;
  const unsigned short* atile = xb + (size_t)rt * 16 * IN_CH;
  v8f aq = {}, ak = {}, av = {}, as2 = {};
#pragma unroll
  for (int kk = 0; kk < IN_CH; kk += 32) {
    v16bf a = load_a_frag(atile, IN_CH, lane, kk);   // reused for 4 weights
    aq  = wmma_bf16(a, load_b_frag(WqT, n0, lane, kk), aq);
    ak  = wmma_bf16(a, load_b_frag(WkT, n0, lane, kk), ak);
    av  = wmma_bf16(a, load_b_frag(WvT, n0, lane, kk), av);
    as2 = wmma_bf16(a, load_b_frag(WsT, n0, lane, kk), as2);
  }
  const size_t ob = (size_t)rt * 16 * OUT_CH;
  store_tile_f32(Q + ob, OUT_CH, aq,  bq, n0, lane);
  store_tile_f32(K + ob, OUT_CH, ak,  bk, n0, lane);
  store_tile_f32(V + ob, OUT_CH, av,  bv, n0, lane);
  store_tile_f32(S + ob, OUT_CH, as2, bs, n0, lane);
}

// ---------- edge tile: build edge_attr (bf16) + e-proj via WMMA ----------
// Ends with s_wait_tensorcnt(0) (no-op for waves without a pending TDM op)
// so any TDM gathers issued at kernel entry are complete at the final
// barrier, fully overlapped with the edge-attr build and the WMMA GEMM.

__device__ __forceinline__ void edge_tile_eproj(
    int g0, const int* __restrict__ src, const int* __restrict__ lastu,
    const int* __restrict__ tarr, const float* __restrict__ msg,
    const float* __restrict__ Wt, const float* __restrict__ bt,
    const unsigned short* __restrict__ WeT,
    unsigned short* ea_tile /*16*128 bf16*/, float* ep_tile /*16*128 f32*/) {
  const int tid = threadIdx.x;
  const int el  = tid >> 4;            // edge within tile (16 threads/edge)
  const int cb  = (tid & 15) * 8;      // 8 consecutive cols
  const int g   = g0 + el;
  const float rel_t = (float)(lastu[src[g]] - tarr[g]);
#pragma unroll
  for (int j = 0; j < 8; ++j) {
    const int c = cb + j;
    float v;
    if (c < TIME_DIM)       v = cosf(fmaf(rel_t, Wt[c], bt[c]));
    else if (c == TIME_DIM) v = msg[g];
    else                    v = 0.0f;          // K padding 101 -> 128
    ea_tile[el * IN_CH + c] = f2bf(v);
  }
  __syncthreads();
  const int wave = tid >> 5, lane = tid & 31;
  const int n0 = wave * 16;
  v8f acc = {};
#pragma unroll
  for (int kk = 0; kk < IN_CH; kk += 32)
    acc = wmma_bf16(load_a_frag(ea_tile, IN_CH, lane, kk),
                    load_b_frag(WeT, n0, lane, kk), acc);
  const int col = n0 + (lane & 15);
  const int rb  = (lane & 16) ? 8 : 0;
#pragma unroll
  for (int r = 0; r < 8; ++r) ep_tile[(rb + r) * OUT_CH + col] = acc[r];
  __builtin_amdgcn_s_wait_tensorcnt(0);
  __syncthreads();
}

// ---------- pass 1: alpha = q.(k+e)/sqrt(32), segment max -----------------

__global__ __launch_bounds__(256)
void edge_pass1_kernel(const int* __restrict__ src, const int* __restrict__ dst,
                       const int* __restrict__ lastu, const int* __restrict__ tarr,
                       const float* __restrict__ msg, const float* __restrict__ Wt,
                       const float* __restrict__ bt, const unsigned short* __restrict__ WeT,
                       const float* __restrict__ Q, const float* __restrict__ K,
                       float* __restrict__ alpha, float* __restrict__ mmax) {
  __shared__ __align__(16) unsigned short ea_tile[16 * IN_CH];
  __shared__ __align__(16) float          ep_tile[16 * OUT_CH];
  __shared__ __align__(16) float          ktile[16 * OUT_CH];
  __shared__ __align__(16) float          qtile[16 * OUT_CH];
  const int g0  = blockIdx.x * 16;
  const int tid = threadIdx.x;
  const int wave = tid >> 5;
  // TDM gathers of the 16 K-rows (by src) and 16 Q-rows (by dst) run on the
  // DMA engine while this block computes edge_attr and the e-proj WMMA.
  if (wave == 0)
    tdm_gather16(K, (unsigned)(uintptr_t)&ktile[0], src, g0, 65535u);
  else if (wave == 1)
    tdm_gather16(Q, (unsigned)(uintptr_t)&qtile[0], dst, g0, 65535u);
  edge_tile_eproj(g0, src, lastu, tarr, msg, Wt, bt, WeT, ea_tile, ep_tile);

  // 64 (edge,head) pairs, 4 threads each (8 dims/thread), wave32 shuffle reduce
  const int pair = tid >> 2, sub = tid & 3;
  const int el = pair >> 2, h = pair & 3;
  const int g  = g0 + el;
  const int d0 = dst[g];
  const float* qr = qtile  + el * OUT_CH + h * HEAD_DIM + sub * 8;
  const float* kr = ktile  + el * OUT_CH + h * HEAD_DIM + sub * 8;
  const float* er = ep_tile + el * OUT_CH + h * HEAD_DIM + sub * 8;
  float p = 0.0f;
#pragma unroll
  for (int j = 0; j < 8; ++j) p = fmaf(qr[j], kr[j] + er[j], p);
  p += __shfl_xor(p, 1, 32);
  p += __shfl_xor(p, 2, 32);
  if (sub == 0) {
    const float a = p * 0.17677669529663687f;   // 1/sqrt(32)
    alpha[(size_t)g * HEADS + h] = a;
    atomicMaxF(&mmax[(size_t)d0 * HEADS + h], a);
  }
}

// ---------- pass 2: exp + segment sum -------------------------------------

__global__ void edge_pass2_kernel(const int* __restrict__ dst,
                                  const float* __restrict__ mmax,
                                  float* __restrict__ alpha,
                                  float* __restrict__ den, int E) {
  int i = blockIdx.x * blockDim.x + threadIdx.x;
  if (i >= E * HEADS) return;
  const int g = i >> 2, h = i & 3;
  const int d0 = dst[g];
  const float ev = __expf(alpha[i] - mmax[(size_t)d0 * HEADS + h]);
  alpha[i] = ev;                                 // alpha buffer now holds ea
  atomicAdd(&den[(size_t)d0 * HEADS + h], ev);
}

// ---------- pass 3: scatter (v+e)*a into agg ------------------------------

__global__ __launch_bounds__(256)
void edge_pass3_kernel(const int* __restrict__ src, const int* __restrict__ dst,
                       const int* __restrict__ lastu, const int* __restrict__ tarr,
                       const float* __restrict__ msg, const float* __restrict__ Wt,
                       const float* __restrict__ bt, const unsigned short* __restrict__ WeT,
                       const float* __restrict__ V, const float* __restrict__ ea,
                       const float* __restrict__ den, float* __restrict__ agg) {
  __shared__ __align__(16) unsigned short ea_tile[16 * IN_CH];
  __shared__ __align__(16) float          ep_tile[16 * OUT_CH];
  __shared__ __align__(16) float          vtile[16 * OUT_CH];
  const int g0  = blockIdx.x * 16;
  const int tid = threadIdx.x;
  if ((tid >> 5) == 0)   // TDM gather of the 16 V-rows (by src), overlapped
    tdm_gather16(V, (unsigned)(uintptr_t)&vtile[0], src, g0, 65535u);
  edge_tile_eproj(g0, src, lastu, tarr, msg, Wt, bt, WeT, ea_tile, ep_tile);

  const int el = tid >> 4;
  const int cb = (tid & 15) * 8;        // 8 cols stay inside one head (8|32)
  const int h  = cb >> 5;
  const int g  = g0 + el;
  const int d0 = dst[g];
  const float a = ea[(size_t)g * HEADS + h] /
                  (den[(size_t)d0 * HEADS + h] + 1e-16f);
  const float* vr = vtile  + el * OUT_CH + cb;
  const float* er = ep_tile + el * OUT_CH + cb;
  float* ar = agg + (size_t)d0 * OUT_CH + cb;
#pragma unroll
  for (int j = 0; j < 8; ++j) atomicAdd(&ar[j], (vr[j] + er[j]) * a);
}

// ---------- finalize: skip + relu + LayerNorm (wave32 per node) -----------

__global__ __launch_bounds__(256)
void finalize_kernel(const float* __restrict__ agg, const float* __restrict__ S,
                     const float* __restrict__ gamma, const float* __restrict__ beta,
                     float* __restrict__ out) {
  const int node = blockIdx.x * 8 + (threadIdx.x >> 5);
  const int lane = threadIdx.x & 31;
  const size_t base = (size_t)node * OUT_CH;
  float v[4], sum = 0.0f, ss = 0.0f;
#pragma unroll
  for (int i = 0; i < 4; ++i) {
    const int c = lane + 32 * i;
    float t0 = agg[base + c] + S[base + c];
    t0 = t0 > 0.0f ? t0 : 0.0f;                  // relu
    v[i] = t0; sum += t0; ss += t0 * t0;
  }
#pragma unroll
  for (int off = 16; off > 0; off >>= 1) {
    sum += __shfl_xor(sum, off, 32);
    ss  += __shfl_xor(ss,  off, 32);
  }
  const float mean = sum * (1.0f / OUT_CH);
  const float var  = ss * (1.0f / OUT_CH) - mean * mean;
  const float inv  = rsqrtf(var + 1e-5f);
#pragma unroll
  for (int i = 0; i < 4; ++i) {
    const int c = lane + 32 * i;
    out[base + c] = (v[i] - mean) * inv * gamma[c] + beta[c];
  }
}

// ---------- launch --------------------------------------------------------

extern "C" void kernel_launch(void* const* d_in, const int* in_sizes, int n_in,
                              void* d_out, int out_size, void* d_ws, size_t ws_size,
                              hipStream_t stream) {
  const float* x     = (const float*)d_in[0];
  const float* msg   = (const float*)d_in[1];
  const float* Wt    = (const float*)d_in[2];
  const float* bt    = (const float*)d_in[3];
  const float* Wq    = (const float*)d_in[4];
  const float* bq    = (const float*)d_in[5];
  const float* Wk    = (const float*)d_in[6];
  const float* bk    = (const float*)d_in[7];
  const float* Wv    = (const float*)d_in[8];
  const float* bv    = (const float*)d_in[9];
  const float* We    = (const float*)d_in[10];
  const float* Wskip = (const float*)d_in[11];
  const float* bskip = (const float*)d_in[12];
  const float* gamma = (const float*)d_in[13];
  const float* beta  = (const float*)d_in[14];
  const int* lastu   = (const int*)d_in[15];
  const int* eidx    = (const int*)d_in[16];
  const int* tarr    = (const int*)d_in[17];

  const int N = in_sizes[0] / IN_CH;    // 50000 (divisible by 16)
  const int E = in_sizes[16] / 2;       // 800000 (divisible by 16)
  const int* src = eidx;
  const int* dst = eidx + E;

  // workspace carve-out (256B-aligned slabs)
  char* w = (char*)d_ws;
  size_t off = 0;
  auto alloc = [&](size_t bytes) -> void* {
    void* p = w + off;
    off += (bytes + 255) & ~(size_t)255;
    return p;
  };
  float* Q   = (float*)alloc((size_t)N * OUT_CH * 4);
  float* K   = (float*)alloc((size_t)N * OUT_CH * 4);
  float* V   = (float*)alloc((size_t)N * OUT_CH * 4);
  float* S   = (float*)alloc((size_t)N * OUT_CH * 4);
  float* agg = (float*)alloc((size_t)N * OUT_CH * 4);
  float* alpha = (float*)alloc((size_t)E * HEADS * 4);
  float* mmax  = (float*)alloc((size_t)N * HEADS * 4);
  float* den   = (float*)alloc((size_t)N * HEADS * 4);
  unsigned short* xb  = (unsigned short*)alloc((size_t)N * IN_CH * 2);
  unsigned short* WqT = (unsigned short*)alloc(128 * 128 * 2);
  unsigned short* WkT = (unsigned short*)alloc(128 * 128 * 2);
  unsigned short* WvT = (unsigned short*)alloc(128 * 128 * 2);
  unsigned short* WsT = (unsigned short*)alloc(128 * 128 * 2);
  unsigned short* WeT = (unsigned short*)alloc(128 * 128 * 2);

  const int nx = N * IN_CH;
  conv_x_kernel<<<(nx + 255) / 256, 256, 0, stream>>>(x, xb, nx);
  conv_w_kernel<<<(5 * 128 * 128 + 255) / 256, 256, 0, stream>>>(
      Wq, Wk, Wv, Wskip, We, WqT, WkT, WvT, WsT, WeT);
  const int ninit = N * OUT_CH + 2 * N * HEADS;
  init_kernel<<<(ninit + 255) / 256, 256, 0, stream>>>(agg, mmax, den, N);

  node_proj_kernel<<<N / 16, 256, 0, stream>>>(xb, WqT, WkT, WvT, WsT,
                                               bq, bk, bv, bskip, Q, K, V, S);
  edge_pass1_kernel<<<E / 16, 256, 0, stream>>>(src, dst, lastu, tarr, msg, Wt, bt,
                                                WeT, Q, K, alpha, mmax);
  edge_pass2_kernel<<<(E * HEADS + 255) / 256, 256, 0, stream>>>(dst, mmax, alpha,
                                                                 den, E);
  edge_pass3_kernel<<<E / 16, 256, 0, stream>>>(src, dst, lastu, tarr, msg, Wt, bt,
                                                WeT, V, alpha, den, agg);
  finalize_kernel<<<N / 8, 256, 0, stream>>>(agg, S, gamma, beta, (float*)d_out);
}
